// StructureModule_40501541601307
// MI455X (gfx1250) — matrix operand order repaired and checked by reading, not compiled, not run on hardware
//
#include <hip/hip_runtime.h>
#include <math.h>

typedef __attribute__((ext_vector_type(16))) _Float16 v16h;
typedef __attribute__((ext_vector_type(8)))  _Float16 v8h;
typedef __attribute__((ext_vector_type(8)))  float    v8f;
typedef __attribute__((ext_vector_type(4)))  unsigned int v4u;
typedef __attribute__((ext_vector_type(4)))  int      v4i;
typedef __attribute__((ext_vector_type(8)))  int      v8i;

#define LL    512
#define CS    256
#define CZ    128
#define NH    8
#define NPROJ 1152   // q(256)+k(256)+v(256)+qp(96)+kp(96)+vp(192)
#define NCAT  1536   // o(256)+opt(192)+optn(64)+opair(1024)

static constexpr float WL_C  = 0.57735026918962576f;  // sqrt(1/3)
static constexpr float WC_C  = 0.23570226039551587f;  // sqrt(2/(9*4))
static constexpr float RSQ_C = 0.17677669529663689f;  // 1/sqrt(32)

#if defined(__has_builtin)
#if __has_builtin(__builtin_amdgcn_tensor_load_to_lds)
#define HAVE_TDM 1
#endif
#endif

// ---------------- WMMA helpers (gfx1250 wave32 layouts) ----------------

static __device__ __forceinline__ v8f wmma_f16(v16h a, v16h b, v8f c) {
  return __builtin_amdgcn_wmma_f32_16x16x32_f16(false, a, false, b, (short)0, c,
                                                false, false);
}

// A tile 16x32 from row-major f16 (M x K). lane&15 = M row.
// Per-lane data is two contiguous 16B chunks: k = 8*hi+e  and  k = 8*hi+16+e.
static __device__ __forceinline__ v16h ld_a_h(const _Float16* A, int lda, int r0,
                                              int k0) {
  int lane = threadIdx.x & 31, m = lane & 15, hi = lane >> 4;
  const _Float16* p = A + (size_t)(r0 + m) * lda + k0 + 8 * hi;
  v8h lo = *(const v8h*)p;
  v8h hi8 = *(const v8h*)(p + 16);
  v16h out;
#pragma unroll
  for (int e = 0; e < 8; ++e) { out[e] = lo[e]; out[e + 8] = hi8[e]; }
  return out;
}

// B tile 32x16 from K-contiguous ("transposed", N x K) f16. lane&15 = N col.
// Per-lane data is two contiguous 16B chunks: k = 8*hi+e  and  k = 8*hi+16+e.
static __device__ __forceinline__ v16h ld_bt_h(const _Float16* Bt, int ldbt,
                                               int n0, int k0) {
  int lane = threadIdx.x & 31, n = lane & 15, hi = lane >> 4;
  const _Float16* p = Bt + (size_t)(n0 + n) * ldbt + k0 + 8 * hi;
  v8h c0 = *(const v8h*)p;
  v8h c1 = *(const v8h*)(p + 16);
  v16h out;
#pragma unroll
  for (int e = 0; e < 8; ++e) { out[e] = c0[e]; out[e + 8] = c1[e]; }
  return out;
}

// --------- TDM: load a [rows x 128] f32 tile of z into LDS ----------
static __device__ __forceinline__ void tdm_issue_z(float* lds_dst,
                                                   const float* gsrc, int rows) {
#ifdef HAVE_TDM
  unsigned lds_off = (unsigned)(size_t)(void*)lds_dst;
  unsigned long long ga = (unsigned long long)(size_t)(const void*)gsrc;
  // D# group0: count=1 | lds_addr | global_addr[56:0] | type=2 (bits 127:126)
  v4u g0 = {1u, lds_off, (unsigned)(ga & 0xffffffffu),
            (unsigned)((ga >> 32) & 0x01ffffffu) | 0x80000000u};
  // D# group1: data_size=4B; tensor_dim0=128, tensor_dim1=512;
  //            tile_dim0=128, tile_dim1=rows; tensor_dim0_stride=128
  v8i g1 = {0x00020000, 0x00800000, 0x02000000, 0x00800000, rows, 128, 0, 0};
  v4i z4 = {0, 0, 0, 0};
#if __clang_major__ >= 23
  v8i z8 = {0, 0, 0, 0, 0, 0, 0, 0};
  __builtin_amdgcn_tensor_load_to_lds(g0, g1, z4, z4, z8, 0);
#else
  __builtin_amdgcn_tensor_load_to_lds(g0, g1, z4, z4, 0);
#endif
#else
  (void)lds_dst; (void)gsrc; (void)rows;
#endif
}

// ---------------- generic WMMA GEMM ----------------
// Out = [resid +] [mask*] act(A*Bt^T + bias);  optional f16 mirror of Out.
template <int RELU, int RES, int MASK, int WH>
__global__ void k_gemm(const _Float16* __restrict__ A, const _Float16* __restrict__ Bt,
                       const float* __restrict__ bias, const float* __restrict__ resid,
                       const float* __restrict__ maskrow, float* __restrict__ Out,
                       _Float16* __restrict__ OutH, int K, int ldA, int ldO) {
  int n0 = blockIdx.x * 16, m0 = blockIdx.y * 16;
  v8f acc = {};
  for (int k0 = 0; k0 < K; k0 += 32) {
    v16h a = ld_a_h(A, ldA, m0, k0);
    v16h b = ld_bt_h(Bt, K, n0, k0);
    acc = wmma_f16(a, b, acc);
  }
  int lane = threadIdx.x & 31, n = lane & 15, hi = lane >> 4;
  float bcol = bias[n0 + n];
#pragma unroll
  for (int r = 0; r < 8; ++r) {
    int m = r + 8 * hi;
    float v = acc[r] + bcol;
    if constexpr (RELU) v = fmaxf(v, 0.0f);
    if constexpr (MASK) v *= (maskrow[m0 + m] > 0.5f) ? 1.0f : 0.0f;
    if constexpr (RES) v += resid[(size_t)(m0 + m) * ldO + n0 + n];
    Out[(size_t)(m0 + m) * ldO + n0 + n] = v;
    if constexpr (WH) OutH[(size_t)(m0 + m) * ldO + n0 + n] = (_Float16)v;
  }
}

// ---------------- one-time setup kernels ----------------

__global__ void k_init(const float* __restrict__ s_in, float* s_buf, _Float16* s_h,
                       float* Rb, float* tb) {
  int idx = blockIdx.x * 256 + threadIdx.x;
  if (idx < LL * CS) {
    float v = s_in[idx];
    s_buf[idx] = v;
    s_h[idx] = (_Float16)v;
  }
  if (idx < LL * 9) {
    int ii = idx % 9;
    Rb[idx] = (ii == 0 || ii == 4 || ii == 8) ? 1.0f : 0.0f;
  }
  if (idx < LL * 3) tb[idx] = 0.0f;
}

// WcatT[n][k]  (transposed, K-contiguous)
__global__ void k_cvt_wcat_t(const float* wq, const float* wk, const float* wv,
                             const float* wqp, const float* wkp, const float* wvp,
                             _Float16* WcatT) {
  int idx = blockIdx.x * 256 + threadIdx.x;
  if (idx >= NPROJ * CS) return;
  int n = idx / CS, kk = idx % CS;
  float v;
  if (n < 256)       v = wq [kk * 256 + n];
  else if (n < 512)  v = wk [kk * 256 + n - 256];
  else if (n < 768)  v = wv [kk * 256 + n - 512];
  else if (n < 864)  v = wqp[kk * 96  + n - 768];
  else if (n < 960)  v = wkp[kk * 96  + n - 864];
  else               v = wvp[kk * 192 + n - 960];
  WcatT[idx] = (_Float16)v;
}

__global__ void k_cvt_bias(const float* bq, const float* bk, const float* bv,
                           const float* bqp, const float* bkp, const float* bvp,
                           float* bias_cat) {
  int n = blockIdx.x * 256 + threadIdx.x;
  if (n >= NPROJ) return;
  float v;
  if (n < 256)       v = bq [n];
  else if (n < 512)  v = bk [n - 256];
  else if (n < 768)  v = bv [n - 512];
  else if (n < 864)  v = bqp[n - 768];
  else if (n < 960)  v = bkp[n - 864];
  else               v = bvp[n - 960];
  bias_cat[n] = v;
}

// dst[n][k] = (f16) src[k][n]   (generic weight transpose K x N -> N x K)
__global__ void k_cvt_t(const float* __restrict__ src, _Float16* __restrict__ dst,
                        int K, int N) {
  int idx = blockIdx.x * 256 + threadIdx.x;
  if (idx >= K * N) return;
  int n = idx / K, kk = idx % K;
  dst[idx] = (_Float16)src[(size_t)kk * N + n];
}

// pairb[h][i][j] = sum_c z[i,j,c]*wpb[c,h] + bpb[h]
__global__ void k_pairbias(const float* __restrict__ z, const float* __restrict__ wpb,
                           const float* __restrict__ bpb, float* __restrict__ pairb) {
  int idx = blockIdx.x * 256 + threadIdx.x;  // i*512 + j
  int i = idx >> 9, j = idx & 511;
  const float* zp = z + (size_t)idx * CZ;
  __builtin_prefetch(zp + 64, 0, 1);  // global_prefetch_b8
  float acc[NH] = {};
  for (int c = 0; c < CZ; ++c) {
    float zv = zp[c];
#pragma unroll
    for (int h = 0; h < NH; ++h) acc[h] += zv * wpb[c * NH + h];
  }
#pragma unroll
  for (int h = 0; h < NH; ++h)
    pairb[((size_t)h * LL + i) * LL + j] = acc[h] + bpb[h];
}

// ---------------- per-block kernels ----------------

// rigid transform of point projections, f16 padded copies, transposed vpg, norms
__global__ void k_rigid(const float* __restrict__ proj, const float* __restrict__ Rb,
                        const float* __restrict__ tb, float* qpg, float* kpg,
                        _Float16* qpgP, _Float16* kpgP, _Float16* vpgT,
                        float* qn, float* kn) {
  int l = blockIdx.x, t = threadIdx.x;  // 128 threads
  float Rm[9], tv[3];
#pragma unroll
  for (int ii = 0; ii < 9; ++ii) Rm[ii] = Rb[l * 9 + ii];
#pragma unroll
  for (int ii = 0; ii < 3; ++ii) tv[ii] = tb[l * 3 + ii];
  const float* row = proj + (size_t)l * NPROJ;
  if (t < 32) {                      // qp: (h = t>>2, p = t&3)
    int h = t >> 2, pp = t & 3;
    const float* src = row + 768 + t * 3;
    float x = src[0], y = src[1], zc = src[2];
#pragma unroll
    for (int i = 0; i < 3; ++i) {
      float g = Rm[i * 3 + 0] * x + Rm[i * 3 + 1] * y + Rm[i * 3 + 2] * zc + tv[i];
      qpg[(size_t)l * 96 + t * 3 + i] = g;
      qpgP[(size_t)l * 256 + h * 32 + pp * 3 + i] = (_Float16)g;
    }
  } else if (t < 64) {               // kp
    int t2 = t - 32, h = t2 >> 2, pp = t2 & 3;
    const float* src = row + 864 + t2 * 3;
    float x = src[0], y = src[1], zc = src[2];
#pragma unroll
    for (int i = 0; i < 3; ++i) {
      float g = Rm[i * 3 + 0] * x + Rm[i * 3 + 1] * y + Rm[i * 3 + 2] * zc + tv[i];
      kpg[(size_t)l * 96 + t2 * 3 + i] = g;
      kpgP[(size_t)l * 256 + h * 32 + pp * 3 + i] = (_Float16)g;
    }
  } else {                           // vp: (h = t3>>3, p = t3&7) -> transposed
    int t3 = t - 64, h = t3 >> 3, pp = t3 & 7;
    const float* src = row + 960 + t3 * 3;
    float x = src[0], y = src[1], zc = src[2];
#pragma unroll
    for (int i = 0; i < 3; ++i) {
      float g = Rm[i * 3 + 0] * x + Rm[i * 3 + 1] * y + Rm[i * 3 + 2] * zc + tv[i];
      vpgT[(size_t)(h * 32 + pp * 3 + i) * LL + l] = (_Float16)g;
    }
  }
  // zero-fill K padding (cols 12..31 per head) of qpgP/kpgP
  for (int idx = t; idx < 320; idx += 128) {
    int which = idx >= 160;
    int rem = idx - which * 160;
    int h = rem / 20, cc = 12 + rem % 20;
    (which ? kpgP : qpgP)[(size_t)l * 256 + h * 32 + cc] = (_Float16)0.0f;
  }
  // zero-fill N padding rows (24..31 per head) of vpgT at column l
  if (t < 64) {
    int h = t >> 3, rr = 24 + (t & 7);
    vpgT[(size_t)(h * 32 + rr) * LL + l] = (_Float16)0.0f;
  }
  __syncthreads();
  if (t < 8) {
    float s = 0.0f;
#pragma unroll
    for (int e = 0; e < 12; ++e) {
      float v = qpg[(size_t)l * 96 + t * 12 + e];
      s += v * v;
    }
    qn[l * 8 + t] = s;
  } else if (t < 16) {
    int h = t - 8;
    float s = 0.0f;
#pragma unroll
    for (int e = 0; e < 12; ++e) {
      float v = kpg[(size_t)l * 96 + h * 12 + e];
      s += v * v;
    }
    kn[l * 8 + h] = s;
  }
}

// vT[h*32+c][j] = proj[j][512 + h*32 + c]  (transposed f16 copy of v)
__global__ void k_vT(const float* __restrict__ proj, _Float16* __restrict__ vT) {
  int idx = blockIdx.x * 256 + threadIdx.x;  // 256*512
  int rr = idx >> 9, j = idx & 511;
  vT[(size_t)rr * LL + j] = (_Float16)proj[(size_t)j * NPROJ + 512 + rr];
}

// logits (2 WMMAs per tile) + softmax -> probs_h[h][i][j] (f16)
__global__ void k_logits(const _Float16* __restrict__ proj_h,
                         const _Float16* __restrict__ qpgP,
                         const _Float16* __restrict__ kpgP,
                         const float* __restrict__ qn, const float* __restrict__ kn,
                         const float* __restrict__ pairb, const float* __restrict__ graw,
                         const float* __restrict__ mask, _Float16* __restrict__ probs_h) {
  __shared__ float lg[16][LL];  // 32 KB
  int h = blockIdx.y, i0 = blockIdx.x * 16;
  int wave = threadIdx.x >> 5, lane = threadIdx.x & 31;
  int n = lane & 15, hi = lane >> 4;
  float gr = graw[h];
  float gamma = (gr > 20.0f) ? gr : log1pf(__expf(gr));
  float gcoef = gamma * (WC_C * 0.5f);

  for (int jt = wave; jt < 32; jt += 8) {
    int j0 = jt * 16;
    v8f sc = {}, cr = {};
    v16h aq = ld_a_h(proj_h, NPROJ, i0, h * 32);            // q tile
    v16h bk = ld_bt_h(proj_h, NPROJ, j0, 256 + h * 32);     // k^T tile
    sc = wmma_f16(aq, bk, sc);
    v16h ap = ld_a_h(qpgP, 256, i0, h * 32);                // qpg (K padded to 32)
    v16h bp = ld_bt_h(kpgP, 256, j0, h * 32);               // kpg^T
    cr = wmma_f16(ap, bp, cr);
#pragma unroll
    for (int r = 0; r < 8; ++r) {
      int m = r + 8 * hi, i = i0 + m, j = j0 + n;
      float d2 = qn[i * 8 + h] + kn[j * 8 + h] - 2.0f * cr[r];
      float lv = WL_C * (sc[r] * RSQ_C + pairb[((size_t)h * LL + i) * LL + j] -
                         gcoef * d2);
      if (mask[j] <= 0.5f) lv = -1e9f;
      lg[m][j] = lv;
    }
  }
  __syncthreads();
  for (int m = wave * 2; m < wave * 2 + 2; ++m) {
    float mx = -3.4e38f;
    for (int j = lane; j < LL; j += 32) mx = fmaxf(mx, lg[m][j]);
#pragma unroll
    for (int o = 16; o >= 1; o >>= 1) mx = fmaxf(mx, __shfl_xor(mx, o));
    float sum = 0.0f;
    for (int j = lane; j < LL; j += 32) {
      float e = __expf(lg[m][j] - mx);
      lg[m][j] = e;
      sum += e;
    }
#pragma unroll
    for (int o = 16; o >= 1; o >>= 1) sum += __shfl_xor(sum, o);
    float rinv = 1.0f / sum;
    for (int j = lane; j < LL; j += 32)
      probs_h[((size_t)h * LL + (i0 + m)) * LL + j] = (_Float16)(lg[m][j] * rinv);
  }
}

// o = a @ v ; optg = a @ vpg   (WMMA over K=512, all-f16 vectorized operands)
__global__ void k_attnout(const _Float16* __restrict__ probs_h,
                          const _Float16* __restrict__ vT,
                          const _Float16* __restrict__ vpgT,
                          float* __restrict__ obuf, float* __restrict__ optg) {
  int h = blockIdx.y, i0 = blockIdx.x * 16;
  const _Float16* ph = probs_h + (size_t)h * LL * LL;
  v8f ao0 = {}, ao1 = {}, ap0 = {}, ap1 = {};
  for (int k0 = 0; k0 < LL; k0 += 32) {
    v16h a = ld_a_h(ph, LL, i0, k0);
    v16h b0 = ld_bt_h(vT, LL, h * 32 + 0, k0);
    v16h b1 = ld_bt_h(vT, LL, h * 32 + 16, k0);
    v16h c0 = ld_bt_h(vpgT, LL, h * 32 + 0, k0);
    v16h c1 = ld_bt_h(vpgT, LL, h * 32 + 16, k0);  // rows 24..31 are zero
    ao0 = wmma_f16(a, b0, ao0);
    ao1 = wmma_f16(a, b1, ao1);
    ap0 = wmma_f16(a, c0, ap0);
    ap1 = wmma_f16(a, c1, ap1);
  }
  int lane = threadIdx.x & 31, n = lane & 15, hi = lane >> 4;
#pragma unroll
  for (int r = 0; r < 8; ++r) {
    int m = r + 8 * hi, i = i0 + m;
    obuf[(size_t)i * 256 + h * 32 + n] = ao0[r];
    obuf[(size_t)i * 256 + h * 32 + 16 + n] = ao1[r];
    optg[(size_t)i * 192 + h * 24 + n] = ap0[r];
    if (n < 8) optg[(size_t)i * 192 + h * 24 + 16 + n] = ap1[r];
  }
}

// opair[i,h,c] = sum_j a[h,i,j] * z[i,j,c]; z streamed through LDS via TDM
__global__ void k_opair(const _Float16* __restrict__ probs_h,
                        const float* __restrict__ z, float* __restrict__ opair) {
  int i = blockIdx.x;
  int c = threadIdx.x & 127, g = threadIdx.x >> 7;  // g: head group of 4
  const _Float16* p0 = probs_h + ((size_t)(g * 4) * LL + i) * LL;
  const size_t HS = (size_t)LL * LL;
  float a0 = 0.f, a1 = 0.f, a2 = 0.f, a3 = 0.f;
#ifdef HAVE_TDM
  __shared__ float zbuf[2][32 * CZ];  // 2 x 16 KB, double buffered
  const float* zi = z + (size_t)i * LL * CZ;
  if (threadIdx.x < 32) tdm_issue_z(&zbuf[0][0], zi, 32);
  for (int ch = 0; ch < 16; ++ch) {
    if (threadIdx.x < 32) __builtin_amdgcn_s_wait_tensorcnt(0);
    __syncthreads();  // tile ch visible; all consumers of ch-1 done
    if (threadIdx.x < 32 && ch + 1 < 16)
      tdm_issue_z(&zbuf[(ch + 1) & 1][0], zi + (size_t)(ch + 1) * 32 * CZ, 32);
    const float* zb = &zbuf[ch & 1][0];
#pragma unroll 4
    for (int jj = 0; jj < 32; ++jj) {
      int j = ch * 32 + jj;
      float zv = zb[jj * CZ + c];
      a0 += (float)p0[j] * zv;
      a1 += (float)p0[j + HS] * zv;
      a2 += (float)p0[j + 2 * HS] * zv;
      a3 += (float)p0[j + 3 * HS] * zv;
    }
    __syncthreads();  // done consuming ch before it is overwritten
  }
#else
  const float* zi = z + (size_t)i * LL * CZ;
  for (int j = 0; j < LL; ++j) {
    __builtin_prefetch(zi + (size_t)(j + 4) * CZ + c, 0, 1);
    float zv = zi[(size_t)j * CZ + c];
    a0 += (float)p0[j] * zv;
    a1 += (float)p0[j + HS] * zv;
    a2 += (float)p0[j + 2 * HS] * zv;
    a3 += (float)p0[j + 3 * HS] * zv;
  }
#endif
  float* dst = opair + (size_t)i * 1024 + g * 512 + c;
  dst[0] = a0; dst[128] = a1; dst[256] = a2; dst[384] = a3;
}

// cat_h = (f16)[o | R^T(optg - t) | ||.|| | opair]
__global__ void k_finalize(const float* __restrict__ obuf, const float* __restrict__ optg,
                           const float* __restrict__ opair, const float* __restrict__ Rb,
                           const float* __restrict__ tb, _Float16* __restrict__ cat_h) {
  int l = blockIdx.x, t = threadIdx.x;
  _Float16* cl = cat_h + (size_t)l * NCAT;
  cl[t] = (_Float16)obuf[(size_t)l * 256 + t];
  if (t < 64) {
    int h = t >> 3, pp = t & 7;
    float Rm[9];
#pragma unroll
    for (int ii = 0; ii < 9; ++ii) Rm[ii] = Rb[l * 9 + ii];
    float g0 = optg[(size_t)l * 192 + h * 24 + pp * 3 + 0] - tb[l * 3 + 0];
    float g1 = optg[(size_t)l * 192 + h * 24 + pp * 3 + 1] - tb[l * 3 + 1];
    float g2 = optg[(size_t)l * 192 + h * 24 + pp * 3 + 2] - tb[l * 3 + 2];
    float o0 = Rm[0] * g0 + Rm[3] * g1 + Rm[6] * g2;  // R^T
    float o1 = Rm[1] * g0 + Rm[4] * g1 + Rm[7] * g2;
    float o2 = Rm[2] * g0 + Rm[5] * g1 + Rm[8] * g2;
    cl[256 + h * 24 + pp * 3 + 0] = (_Float16)o0;
    cl[256 + h * 24 + pp * 3 + 1] = (_Float16)o1;
    cl[256 + h * 24 + pp * 3 + 2] = (_Float16)o2;
    cl[448 + h * 8 + pp] = (_Float16)sqrtf(o0 * o0 + o1 * o1 + o2 * o2 + 1e-8f);
  }
#pragma unroll
  for (int cc = t; cc < 1024; cc += 256)
    cl[512 + cc] = (_Float16)opair[(size_t)l * 1024 + cc];
}

// backbone update
__global__ void k_backbone(const float* __restrict__ s_buf, const float* __restrict__ wbb,
                           const float* __restrict__ bbb, const float* __restrict__ mask,
                           float* __restrict__ Rb, float* __restrict__ tb) {
  int l = blockIdx.x * 64 + threadIdx.x;
  if (l >= LL) return;
  const float* sr = s_buf + (size_t)l * CS;
  float u[6];
#pragma unroll
  for (int c = 0; c < 6; ++c) {
    float acc = bbb[c];
    for (int kk = 0; kk < CS; ++kk) acc += sr[kk] * wbb[kk * 6 + c];
    u[c] = acc;
  }
  float b = u[0], c = u[1], d = u[2];
  float nq = sqrtf(1.0f + b * b + c * c + d * d);
  float a = 1.0f / nq;
  b /= nq; c /= nq; d /= nq;
  float dR[9];
  dR[0] = a * a + b * b - c * c - d * d; dR[1] = 2 * (b * c - a * d); dR[2] = 2 * (b * d + a * c);
  dR[3] = 2 * (b * c + a * d); dR[4] = a * a - b * b + c * c - d * d; dR[5] = 2 * (c * d - a * b);
  dR[6] = 2 * (b * d - a * c); dR[7] = 2 * (c * d + a * b); dR[8] = a * a - b * b - c * c + d * d;
  float mk = (mask[l] > 0.5f) ? 1.0f : 0.0f;
  float dt[3] = {u[3] * mk, u[4] * mk, u[5] * mk};
  float Rm[9];
#pragma unroll
  for (int ii = 0; ii < 9; ++ii) Rm[ii] = Rb[l * 9 + ii];
#pragma unroll
  for (int i = 0; i < 3; ++i)
    tb[l * 3 + i] += Rm[i * 3 + 0] * dt[0] + Rm[i * 3 + 1] * dt[1] + Rm[i * 3 + 2] * dt[2];
  float Rn[9];
#pragma unroll
  for (int i = 0; i < 3; ++i)
#pragma unroll
    for (int k = 0; k < 3; ++k)
      Rn[i * 3 + k] = Rm[i * 3 + 0] * dR[0 * 3 + k] + Rm[i * 3 + 1] * dR[1 * 3 + k] +
                      Rm[i * 3 + 2] * dR[2 * 3 + k];
#pragma unroll
  for (int ii = 0; ii < 9; ++ii) Rb[l * 9 + ii] = Rn[ii];
}

__global__ void k_writeout(const float* __restrict__ s_buf, const float* __restrict__ Rb,
                           const float* __restrict__ tb, float* __restrict__ out) {
  int idx = blockIdx.x * 256 + threadIdx.x;
  if (idx < LL * CS) out[idx] = s_buf[idx];
  else if (idx < LL * CS + LL * 9) out[idx] = Rb[idx - LL * CS];
  else if (idx < LL * CS + LL * 9 + LL * 3) out[idx] = tb[idx - LL * CS - LL * 9];
}

// ---------------- host launch ----------------

extern "C" void kernel_launch(void* const* d_in, const int* in_sizes, int n_in,
                              void* d_out, int out_size, void* d_ws, size_t ws_size,
                              hipStream_t stream) {
  const float* s_in = (const float*)d_in[0];
  const float* z_in = (const float*)d_in[1];
  const float* mask = (const float*)d_in[2];
  const float* wq = (const float*)d_in[3];   const float* bq = (const float*)d_in[4];
  const float* wk = (const float*)d_in[5];   const float* bk = (const float*)d_in[6];
  const float* wv = (const float*)d_in[7];   const float* bv = (const float*)d_in[8];
  const float* wqp = (const float*)d_in[9];  const float* bqp = (const float*)d_in[10];
  const float* wkp = (const float*)d_in[11]; const float* bkp = (const float*)d_in[12];
  const float* wvp = (const float*)d_in[13]; const float* bvp = (const float*)d_in[14];
  const float* wpb = (const float*)d_in[15]; const float* bpb = (const float*)d_in[16];
  const float* graw = (const float*)d_in[17];
  const float* wo = (const float*)d_in[18];  const float* bo = (const float*)d_in[19];
  const float* wt1 = (const float*)d_in[20]; const float* bt1 = (const float*)d_in[21];
  const float* wt2 = (const float*)d_in[22]; const float* bt2 = (const float*)d_in[23];
  const float* wbb = (const float*)d_in[24]; const float* bbb = (const float*)d_in[25];
  (void)in_sizes; (void)n_in; (void)out_size; (void)ws_size;

  char* p = (char*)d_ws;
  auto alloc = [&](size_t bytes) -> char* {
    char* r = p;
    p += (bytes + 255) & ~(size_t)255;
    return r;
  };
  _Float16* WcatT = (_Float16*)alloc((size_t)NPROJ * CS * 2);
  _Float16* woT   = (_Float16*)alloc((size_t)CS * NCAT * 2);
  _Float16* wt1T  = (_Float16*)alloc((size_t)1024 * CS * 2);
  _Float16* wt2T  = (_Float16*)alloc((size_t)CS * 1024 * 2);
  float* bias_cat = (float*)alloc(NPROJ * 4);
  float* pairb    = (float*)alloc((size_t)NH * LL * LL * 4);
  float* s_buf    = (float*)alloc((size_t)LL * CS * 4);
  _Float16* s_h   = (_Float16*)alloc((size_t)LL * CS * 2);
  float* Rb       = (float*)alloc((size_t)LL * 9 * 4);
  float* tb       = (float*)alloc((size_t)LL * 3 * 4);
  float* proj     = (float*)alloc((size_t)LL * NPROJ * 4);
  _Float16* proj_h= (_Float16*)alloc((size_t)LL * NPROJ * 2);
  float* qpg      = (float*)alloc((size_t)LL * 96 * 4);
  float* kpg      = (float*)alloc((size_t)LL * 96 * 4);
  _Float16* qpgP  = (_Float16*)alloc((size_t)LL * 256 * 2);
  _Float16* kpgP  = (_Float16*)alloc((size_t)LL * 256 * 2);
  _Float16* vT    = (_Float16*)alloc((size_t)256 * LL * 2);
  _Float16* vpgT  = (_Float16*)alloc((size_t)256 * LL * 2);
  float* qn       = (float*)alloc((size_t)LL * 8 * 4);
  float* kn       = (float*)alloc((size_t)LL * 8 * 4);
  _Float16* probs_h = (_Float16*)alloc((size_t)NH * LL * LL * 2);
  float* obuf     = (float*)alloc((size_t)LL * 256 * 4);
  float* optg     = (float*)alloc((size_t)LL * 192 * 4);
  float* opair    = (float*)alloc((size_t)LL * 1024 * 4);
  _Float16* cat_h = (_Float16*)alloc((size_t)LL * NCAT * 2);
  float* tr1      = (float*)alloc((size_t)LL * 1024 * 4);
  _Float16* tr1_h = (_Float16*)alloc((size_t)LL * 1024 * 2);

  // one-time setup
  k_init<<<512, 256, 0, stream>>>(s_in, s_buf, s_h, Rb, tb);
  k_cvt_wcat_t<<<(NPROJ * CS + 255) / 256, 256, 0, stream>>>(wq, wk, wv, wqp, wkp, wvp, WcatT);
  k_cvt_bias<<<(NPROJ + 255) / 256, 256, 0, stream>>>(bq, bk, bv, bqp, bkp, bvp, bias_cat);
  k_cvt_t<<<(NCAT * CS + 255) / 256, 256, 0, stream>>>(wo, woT, NCAT, CS);
  k_cvt_t<<<(CS * 1024 + 255) / 256, 256, 0, stream>>>(wt1, wt1T, CS, 1024);
  k_cvt_t<<<(1024 * CS + 255) / 256, 256, 0, stream>>>(wt2, wt2T, 1024, CS);
  k_pairbias<<<(LL * LL) / 256, 256, 0, stream>>>(z_in, wpb, bpb, pairb);

  for (int blk = 0; blk < 8; ++blk) {
    // proj = s @ Wcat + bias   (+ f16 mirror)
    k_gemm<0, 0, 0, 1><<<dim3(NPROJ / 16, LL / 16), 32, 0, stream>>>(
        s_h, WcatT, bias_cat, nullptr, nullptr, proj, proj_h, CS, CS, NPROJ);
    k_rigid<<<LL, 128, 0, stream>>>(proj, Rb, tb, qpg, kpg, qpgP, kpgP, vpgT, qn, kn);
    k_vT<<<(256 * LL) / 256, 256, 0, stream>>>(proj, vT);
    k_logits<<<dim3(LL / 16, NH), 256, 0, stream>>>(proj_h, qpgP, kpgP, qn, kn,
                                                    pairb, graw, mask, probs_h);
    k_attnout<<<dim3(LL / 16, NH), 32, 0, stream>>>(probs_h, vT, vpgT, obuf, optg);
    k_opair<<<LL, 256, 0, stream>>>(probs_h, z_in, opair);
    k_finalize<<<LL, 256, 0, stream>>>(obuf, optg, opair, Rb, tb, cat_h);
    // s = s + cat @ wo + bo
    k_gemm<0, 1, 0, 1><<<dim3(CS / 16, LL / 16), 32, 0, stream>>>(
        cat_h, woT, bo, s_buf, nullptr, s_buf, s_h, NCAT, NCAT, CS);
    // tr1 = relu(s @ wt1 + bt1)
    k_gemm<1, 0, 0, 1><<<dim3(1024 / 16, LL / 16), 32, 0, stream>>>(
        s_h, wt1T, bt1, nullptr, nullptr, tr1, tr1_h, CS, CS, 1024);
    // s = s + mask * (tr1 @ wt2 + bt2)
    k_gemm<0, 1, 1, 1><<<dim3(CS / 16, LL / 16), 32, 0, stream>>>(
        tr1_h, wt2T, bt2, s_buf, mask, s_buf, s_h, 1024, 1024, CS);
    k_backbone<<<LL / 64, 64, 0, stream>>>(s_buf, wbb, bbb, mask, Rb, tb);
  }
  k_writeout<<<(LL * CS + LL * 12 + 255) / 256, 256, 0, stream>>>(s_buf, Rb, tb,
                                                                  (float*)d_out);
}
